// SelectAttachmentClusters_82489141887283
// MI455X (gfx1250) — matrix operand order repaired and checked by reading, not compiled, not run on hardware
//
#include <hip/hip_runtime.h>
#include <hip/hip_bf16.h>

// ---------------------------------------------------------------------------
// MI455X (gfx1250) fused kernel for:
//   gathered = motifs[batch_indices]; x = concat(node_hiddens, gathered)
//   out = sigmoid(relu(x @ W1 + b1) @ W2 + b2)
//
// bf16 WMMA GEMM (131 GFLOP), W1 packed to fragment order and held in LDS
// (256 KB of the 320 KB WGP LDS), loaded via the Tensor Data Mover when the
// builtin is available. 512-thread blocks (16 waves, 4/SIMD) for latency
// hiding; A-operand raw data double-buffered across the K loop.
// ---------------------------------------------------------------------------

typedef __attribute__((ext_vector_type(16))) __bf16       v16bf;
typedef __attribute__((ext_vector_type(8)))  float        v8f;
typedef __attribute__((ext_vector_type(8)))  unsigned int v8u;
typedef __attribute__((ext_vector_type(4)))  unsigned int v4u;
typedef __attribute__((ext_vector_type(8)))  int          v8i;
typedef __attribute__((ext_vector_type(4)))  int          v4i;

#define DN       256           // D_NODE == D_MOTIF == D_HID
#define DIN      512           // concat width (K)
#define TILE_M   128           // rows per block-iteration
#define BLOCK    512           // 16 waves
#define W1_FRAG_DWORDS 65536   // 512*256 bf16 = 256 KB

__device__ __forceinline__ unsigned int bf16_rne(float f) {
  union { float f; unsigned int u; } c; c.f = f;
  return (c.u + 0x7FFFu + ((c.u >> 16) & 1u)) >> 16;
}

// Pack two f32 -> packed bf16x2 (low = first arg).
#if defined(__has_builtin)
#if __has_builtin(__builtin_amdgcn_cvt_pk_bf16_f32)
#define HAVE_CVT_PK_BF16 1
#endif
#endif

__device__ __forceinline__ unsigned int pack_bf16x2(float lo, float hi) {
#ifdef HAVE_CVT_PK_BF16
  typedef __attribute__((ext_vector_type(2))) __bf16 v2bf;
  v2bf r = __builtin_amdgcn_cvt_pk_bf16_f32(lo, hi);
  return __builtin_bit_cast(unsigned int, r);
#else
  // round-half-up + v_perm_b32 byte-select of the two high halves (3 VALU)
  union { float f; unsigned int u; } a, b; a.f = lo; b.f = hi;
  return __builtin_amdgcn_perm(b.u + 0x8000u, a.u + 0x8000u, 0x07060302u);
#endif
}

// A-fragment (16x32 bf16, ISA 7.12.2): lane l (r16=l&15, hi=l>>4) holds row
// M=r16, K = (v<4?0:16) + hi*8 + (v%4)*2 + h.
__device__ __forceinline__ void load_raw(const float* __restrict__ s, int hi,
                                         float4* __restrict__ r) {
  const float4* p0 = (const float4*)(s + hi * 8);
  const float4* p1 = (const float4*)(s + 16 + hi * 8);
  r[0] = p0[0]; r[1] = p0[1]; r[2] = p1[0]; r[3] = p1[1];
}

__device__ __forceinline__ v16bf pack_frag(const float4* __restrict__ r) {
  v8u pk;
  pk[0] = pack_bf16x2(r[0].x, r[0].y);
  pk[1] = pack_bf16x2(r[0].z, r[0].w);
  pk[2] = pack_bf16x2(r[1].x, r[1].y);
  pk[3] = pack_bf16x2(r[1].z, r[1].w);
  pk[4] = pack_bf16x2(r[2].x, r[2].y);
  pk[5] = pack_bf16x2(r[2].z, r[2].w);
  pk[6] = pack_bf16x2(r[3].x, r[3].y);
  pk[7] = pack_bf16x2(r[3].z, r[3].w);
  return __builtin_bit_cast(v16bf, pk);
}

// ---------------------------------------------------------------------------
// Prep: repack W1 [512,256] f32 -> bf16 WMMA-B fragment order.
// B (32x16): lane l holds col N=l&15; VGPR v holds K pair
// (v<4?0:16) + (l>>4)*8 + (v%4)*2 + {0,1}. Packed bf16 address:
// (((kt*16+nt)*32 + lane)*8 + v)*2 + h  -> 1 KB per fragment, coalesced.
// ---------------------------------------------------------------------------
__global__ void pack_w1_kernel(const float* __restrict__ W1,
                               unsigned short* __restrict__ w1p) {
  int p = blockIdx.x * blockDim.x + threadIdx.x;
  if (p >= DIN * DN) return;
  int h    = p & 1;
  int v    = (p >> 1) & 7;
  int lane = (p >> 4) & 31;
  int frag = p >> 9;
  int nt   = frag & 15;
  int kt   = frag >> 4;
  int K    = kt * 32 + (v < 4 ? 0 : 16) + (lane >> 4) * 8 + (v & 3) * 2 + h;
  int Ncol = nt * 16 + (lane & 15);
  w1p[p] = (unsigned short)bf16_rne(W1[K * DN + Ncol]);
}

// ---------------------------------------------------------------------------
// Main fused kernel: 512 threads = 16 waves. Wave w: mgroup=w&3 (32 rows),
// ngroup=w>>2 in 0..3 (64 hidden cols). acc = 2 M-subtiles x 4 N-tiles.
// ---------------------------------------------------------------------------
__global__ __launch_bounds__(BLOCK)
void fused_mlp_kernel(const float* __restrict__ nh,
                      const float* __restrict__ motifs,
                      const long long* __restrict__ bidx,
                      const unsigned short* __restrict__ w1p,
                      const float* __restrict__ b1,
                      const float* __restrict__ w2,
                      const float* __restrict__ b2,
                      float* __restrict__ out,
                      int n, int ntiles) {
  extern __shared__ unsigned char smem[];
  unsigned int* w1lds = (unsigned int*)smem;                 // 256 KB packed W1
  float* hsum = (float*)(smem + W1_FRAG_DWORDS * 4);         // 128 f32 row sums

  const int tid    = threadIdx.x;
  const int lane   = tid & 31;
  const int wave   = tid >> 5;
  const int r16    = lane & 15;
  const int hi     = lane >> 4;
  const int mgroup = wave & 3;
  const int ngroup = wave >> 2;

  // ---- Stage packed W1 into LDS once per block ----------------------------
#if defined(__has_builtin) && __has_builtin(__builtin_amdgcn_tensor_load_to_lds)
  if (wave == 0) {
    // D# for a linear 256 KB tile: data_size=8B, tile_dim0=tensor_dim0=32768.
    unsigned long long ga = (unsigned long long)(size_t)w1p;
    v4u g0;
    g0[0] = 1u;                                   // count=1 (valid user D#)
    g0[1] = 0u;                                   // lds_addr = 0 (dyn LDS base)
    g0[2] = (unsigned int)(ga & 0xFFFFFFFFu);     // global_addr[31:0]
    g0[3] = (unsigned int)((ga >> 32) & 0x1FFFFFFu) | (2u << 30); // type=2
    v8i g1;
    g1[0] = (int)(3u << 16);        // workgroup_mask=0, data_size=3 (8 bytes)
    g1[1] = (int)(0x8000u << 16);   // tensor_dim0[15:0]=32768 (bits 79:48)
    g1[2] = (int)(1u << 16);        // tensor_dim0 MSBs=0; tensor_dim1=1
    g1[3] = (int)(0x8000u << 16);   // tensor_dim1 MSBs=0; tile_dim0=32768
    g1[4] = 1;                      // tile_dim1=1, tile_dim2=0
    g1[5] = 0x8000;                 // tensor_dim0_stride[31:0]=32768
    g1[6] = 0;                      // stride MSBs / tensor_dim1_stride lo
    g1[7] = 0;
    v4i z = {0, 0, 0, 0};
#if defined(__clang_major__) && (__clang_major__ >= 23)
    v8i z8 = {0, 0, 0, 0, 0, 0, 0, 0};
    __builtin_amdgcn_tensor_load_to_lds(g0, g1, z, z, z8, 0);
#else
    __builtin_amdgcn_tensor_load_to_lds(g0, g1, z, z, 0);
#endif
    __builtin_amdgcn_s_wait_tensorcnt(0);
  }
#else
  {
    const uint4* src = (const uint4*)w1p;
    uint4* dst = (uint4*)w1lds;
    for (int i = tid; i < W1_FRAG_DWORDS / 4; i += BLOCK) dst[i] = src[i];
  }
#endif
  __syncthreads();

  // Per-lane bias / W2 slices for this wave's 64 hidden columns.
  float b1v[4], w2v[4];
#pragma unroll
  for (int nt = 0; nt < 4; ++nt) {
    int col = ngroup * 64 + nt * 16 + r16;
    b1v[nt] = b1[col];
    w2v[nt] = w2[col];
  }
  const float b2s = b2[0];

  for (int tile = blockIdx.x; tile < ntiles; tile += gridDim.x) {
    if (tid < TILE_M) hsum[tid] = 0.0f;
    __syncthreads();

    const int rowbase = tile * TILE_M + mgroup * 32;
    int row0 = rowbase + r16;
    int row1 = rowbase + 16 + r16;
    int row0c = row0 < n ? row0 : n - 1;   // clamp tail rows (stores guarded)
    int row1c = row1 < n ? row1 : n - 1;
    const int g0i = (int)bidx[row0c];
    const int g1i = (int)bidx[row1c];
    const float* nh0 = nh + (size_t)row0c * DN;
    const float* nh1 = nh + (size_t)row1c * DN;
    const float* mt0 = motifs + (size_t)g0i * DN;
    const float* mt1 = motifs + (size_t)g1i * DN;

    v8f acc[2][4];
#pragma unroll
    for (int m = 0; m < 2; ++m)
#pragma unroll
      for (int nt = 0; nt < 4; ++nt) acc[m][nt] = {};

    // K loop, fully unrolled, with A raw data double-buffered one kt ahead
    // so HBM/L2 load latency overlaps the WMMA stream.
    float4 c0[4], c1[4], p0[4], p1[4];
    load_raw(nh0, hi, c0);
    load_raw(nh1, hi, c1);
#pragma unroll
    for (int kt = 0; kt < 16; ++kt) {
      if (kt + 1 < 16) {
        const float* s0 = (kt + 1 < 8) ? (nh0 + (kt + 1) * 32) : (mt0 + (kt - 7) * 32);
        const float* s1 = (kt + 1 < 8) ? (nh1 + (kt + 1) * 32) : (mt1 + (kt - 7) * 32);
        load_raw(s0, hi, p0);
        load_raw(s1, hi, p1);
      }
      v16bf a0 = pack_frag(c0);
      v16bf a1 = pack_frag(c1);
      const unsigned int fragbase = (unsigned int)(kt * 16 + ngroup * 4) * 256;
#pragma unroll
      for (int nt = 0; nt < 4; ++nt) {
        const v16bf bf =
            *(const v16bf*)(w1lds + fragbase + (unsigned int)nt * 256 + lane * 8);
        acc[0][nt] = __builtin_amdgcn_wmma_f32_16x16x32_bf16(
            false, a0, false, bf, (short)0, acc[0][nt], false, false);
        acc[1][nt] = __builtin_amdgcn_wmma_f32_16x16x32_bf16(
            false, a1, false, bf, (short)0, acc[1][nt], false, false);
      }
#pragma unroll
      for (int q = 0; q < 4; ++q) { c0[q] = p0[q]; c1[q] = p1[q]; }
    }

    // Epilogue: bias + ReLU + dot with W2 over this wave's 64 cols, 16-lane
    // half reductions, ds_add_f32 into the shared per-row accumulator.
    // C layout: VGPR r, lanes 0-15 -> M=r, lanes 16-31 -> M=r+8; N=lane&15.
#pragma unroll
    for (int m = 0; m < 2; ++m) {
#pragma unroll
      for (int r = 0; r < 8; ++r) {
        float s = 0.0f;
#pragma unroll
        for (int nt = 0; nt < 4; ++nt) {
          float hv = acc[m][nt][r] + b1v[nt];
          hv = fmaxf(hv, 0.0f);
          s += hv * w2v[nt];
        }
        s += __shfl_xor(s, 1, 32);
        s += __shfl_xor(s, 2, 32);
        s += __shfl_xor(s, 4, 32);
        s += __shfl_xor(s, 8, 32);
        if (r16 == 0) {
          int rowl = mgroup * 32 + m * 16 + r + (hi << 3);
          atomicAdd(&hsum[rowl], s);
        }
      }
    }
    __syncthreads();

    int orow = tile * TILE_M + tid;
    if (tid < TILE_M && orow < n) {
      float x = hsum[tid] + b2s;
      out[orow] = 1.0f / (1.0f + __expf(-x));
    }
    __syncthreads();
  }
}

// ---------------------------------------------------------------------------
extern "C" void kernel_launch(void* const* d_in, const int* in_sizes, int n_in,
                              void* d_out, int out_size, void* d_ws, size_t ws_size,
                              hipStream_t stream) {
  const float*      nh     = (const float*)d_in[0];       // [N,256] f32
  const float*      motifs = (const float*)d_in[1];       // [B,256] f32
  const long long*  bidx   = (const long long*)d_in[2];   // [N] int64
  const float*      W1     = (const float*)d_in[3];       // [512,256] f32
  const float*      b1     = (const float*)d_in[4];       // [256] f32
  const float*      W2     = (const float*)d_in[5];       // [256,1] f32
  const float*      b2     = (const float*)d_in[6];       // [1] f32
  float*            out    = (float*)d_out;               // [N] f32
  unsigned short*   w1p    = (unsigned short*)d_ws;       // 256 KB packed W1

  const int n      = in_sizes[0] / DN;
  const int ntiles = (n + TILE_M - 1) / TILE_M;

  pack_w1_kernel<<<(DIN * DN + 255) / 256, 256, 0, stream>>>(W1, w1p);

  const size_t shmem = (size_t)W1_FRAG_DWORDS * 4 + TILE_M * sizeof(float);
  int grid = ntiles < 1024 ? ntiles : 1024;
  fused_mlp_kernel<<<grid, BLOCK, shmem, stream>>>(nh, motifs, bidx, w1p,
                                                   b1, W2, b2, out, n, ntiles);
}
